// PositionalNgramMemoryNetwork_80109730005689
// MI455X (gfx1250) — compile-verified, matrix operand order checked by reading, not run on hardware
//
#include <hip/hip_runtime.h>
#include <stdint.h>

// ---------- types ----------
typedef __attribute__((ext_vector_type(4)))  unsigned int  u32x4;
typedef __attribute__((ext_vector_type(2)))  unsigned int  u32x2;
typedef __attribute__((ext_vector_type(4)))  float         f32x4;
typedef __attribute__((ext_vector_type(8)))  float         f32x8;
typedef __attribute__((ext_vector_type(16))) __bf16        bf16x16;

struct Frag32 { u32x4 lo, hi; };   // 32 bytes = one 16-bf16 WMMA operand per lane

__device__ __forceinline__ unsigned short f2bf(float f) {
  unsigned int u = __float_as_uint(f);
  return (unsigned short)((u + 0x7FFFu + ((u >> 16) & 1u)) >> 16);  // RNE
}

__device__ __forceinline__ bf16x16 make_frag(const unsigned short* p0,
                                             const unsigned short* p1) {
  Frag32 f;
  f.lo = *(const u32x4*)p0;   // K block {k..k+7}
  f.hi = *(const u32x4*)p1;   // K block {k+16..k+23}
  return __builtin_bit_cast(bf16x16, f);
}

// Problem constants: B=8, L=4096, D=1024, S=256, N=3
#define LL 4096
#define DD 1024

// ---------- prep: memory (S,N,D) f32 -> B^T bf16 [n*256+s][d] ----------
__global__ __launch_bounds__(256) void pnm_convert(const float* __restrict__ mem,
                                                   unsigned short* __restrict__ Bmat) {
  int idx = blockIdx.x * 256 + threadIdx.x;
  if (idx >= 768 * 1024) return;
  int c = idx >> 10;            // 0..767  (c = n*256 + s)
  int d = idx & 1023;
  int n = c >> 8;
  int s = c & 255;
  Bmat[idx] = f2bf(mem[((size_t)(s * 3 + n) << 10) + d]);
}

// X-tile staging helpers (34 rows x 64 cols per chunk, 544 float4 groups)
__device__ __forceinline__ f32x4 load_xg(const float* __restrict__ x,
                                         int gRow0, int l0, int k0, int idx) {
  int row = idx >> 4;              // 0..33
  int c4  = idx & 15;              // group of 4 floats
  int gl  = l0 + row - 2;          // within-batch source row
  f32x4 f = {0.f, 0.f, 0.f, 0.f};  // zero pad before batch start
  if (gl >= 0)
    f = *(const f32x4*)(x + ((size_t)(gRow0 + row - 2) << 10) + k0 + (c4 << 2));
  return f;
}

__device__ __forceinline__ void store_xg(unsigned short* lds_x, int idx, f32x4 f) {
  int row = idx >> 4;
  int c4  = idx & 15;
  unsigned int lo = (unsigned)f2bf(f.x) | ((unsigned)f2bf(f.y) << 16);
  unsigned int hi = (unsigned)f2bf(f.z) | ((unsigned)f2bf(f.w) << 16);
  u32x2 v = {lo, hi};
  *(u32x2*)&lds_x[row * 72 + (c4 << 2)] = v;
}

// ---------- main: shifted GEMMs (bf16 WMMA) + argmax + gather ----------
__global__ __launch_bounds__(256) void pnm_main(const float* __restrict__ x,
                                                const float* __restrict__ mem,
                                                const float* __restrict__ pb,
                                                const unsigned short* __restrict__ Bmat,
                                                float* __restrict__ out) {
  __shared__ __align__(16) unsigned short lds_x[34 * 72];  // 32+2 halo rows, 64 bf16 + pad
  __shared__ float lds_sco[16 * 772];                      // one 16-row half of scores
  __shared__ float lds_bias[768];                          // bias[n*256+s]
  __shared__ int   lds_best[48];                           // best[row*3+n]

  const int tid   = threadIdx.x;
  const int lid   = tid & 31;          // lane in wave32
  const int wv    = tid >> 5;          // wave 0..7
  const int rsel  = lid >> 4;          // lane half selects K sub-block
  const int rlane = lid & 15;          // row (A) / col (B) within 16-tile

  const int gRow0 = blockIdx.x << 5;   // flat row over B*L (tiles never cross batches)
  const int l0    = gRow0 & (LL - 1);  // row within batch

  // stage positional bias: lds_bias[n*256+s] = pb[s*3+n]
  for (int idx = tid; idx < 768; idx += 256) {
    int n = idx >> 8, s = idx & 255;
    lds_bias[idx] = pb[s * 3 + n];
  }

  // Column tile mapping: ct = wv + 8*j  =>  n(ct) = j>>1, compile-time per unrolled j.
  const unsigned short* bbase[6];
#pragma unroll
  for (int j = 0; j < 6; ++j) {
    int ct = wv + 8 * j;
    bbase[j] = Bmat + ((size_t)((ct << 4) + rlane) << 10);
  }

  f32x8 acc[6][2] = {};   // [col tile j][row tile r]; 32 rows x 96 cols per wave

  // prologue: prefetch first X chunk into registers
  f32x4 xr0 = load_xg(x, gRow0, l0, 0, tid);
  f32x4 xr1 = load_xg(x, gRow0, l0, 0, tid + 256);
  f32x4 xr2 = (tid < 32) ? load_xg(x, gRow0, l0, 0, tid + 512) : f32x4{0.f, 0.f, 0.f, 0.f};

  for (int k0 = 0; k0 < DD; k0 += 64) {
    // ---- convert + store staged chunk to LDS ----
    store_xg(lds_x, tid, xr0);
    store_xg(lds_x, tid + 256, xr1);
    if (tid < 32) store_xg(lds_x, tid + 512, xr2);
    __syncthreads();

    // ---- prefetch next chunk (latency hides under WMMA work below) ----
    if (k0 + 64 < DD) {
      xr0 = load_xg(x, gRow0, l0, k0 + 64, tid);
      xr1 = load_xg(x, gRow0, l0, k0 + 64, tid + 256);
      if (tid < 32) xr2 = load_xg(x, gRow0, l0, k0 + 64, tid + 512);
    }

    // ---- two K=32 WMMA steps per 64-wide chunk ----
#pragma unroll
    for (int ksub = 0; ksub < 2; ++ksub) {
      int eb = ksub * 32 + rsel * 8;
      // 6 A fragments: 3 n-gram shifts x 2 row tiles, all from one LDS tile
      bf16x16 aF[3][2];
#pragma unroll
      for (int n = 0; n < 3; ++n) {
#pragma unroll
        for (int r = 0; r < 2; ++r) {
          const unsigned short* p = &lds_x[(r * 16 + n + rlane) * 72 + eb];
          aF[n][r] = make_frag(p, p + 16);
        }
      }

      int kk = k0 + ksub * 32 + rsel * 8;
      // 6 B fragments, each feeds two WMMAs (both row tiles)
      bf16x16 bfr[6];
#pragma unroll
      for (int j = 0; j < 6; ++j)
        bfr[j] = make_frag(bbase[j] + kk, bbase[j] + kk + 16);

#pragma unroll
      for (int j = 0; j < 6; ++j) {
#pragma unroll
        for (int r = 0; r < 2; ++r) {
          acc[j][r] = __builtin_amdgcn_wmma_f32_16x16x32_bf16(
              false, aF[j >> 1][r], false, bfr[j], (short)0, acc[j][r], false, false);
        }
      }
    }
    __syncthreads();
  }

  // ---- two-phase epilogue: row tiles 0 and 1 share the 16-row score buffer ----
#pragma unroll 1
  for (int half = 0; half < 2; ++half) {
    // scores -> LDS (D layout: VGPR g = row g / g+8, lane = col)
#pragma unroll
    for (int j = 0; j < 6; ++j) {
      int ct  = wv + 8 * j;
      int col = (ct << 4) + rlane;
#pragma unroll
      for (int g = 0; g < 8; ++g) {
        int row = g + (rsel << 3);
        lds_sco[row * 772 + col] = acc[j][half][g];
      }
    }
    __syncthreads();   // scores visible; also fences previous phase's gather vs argmax

    // per (row, n) argmax over 256 slots (first-max tie rule like jnp.argmax)
    if (tid < 48) {
      int row = tid & 15;
      int n   = tid >> 4;
      const float* sp = &lds_sco[row * 772 + (n << 8)];
      const float* bb = &lds_bias[n << 8];
      float best = sp[0] + bb[0];
      int bi = 0;
      for (int s = 1; s < 256; ++s) {
        float v = sp[s] + bb[s];
        if (v > best) { best = v; bi = s; }
      }
      lds_best[row * 3 + n] = bi;
    }
    __syncthreads();

    // gather winning memory rows (f32, L2-resident) and sum over n
    int g = tid;   // 256 float4 groups cover D=1024
    int rbase = gRow0 + half * 16;
    for (int row = 0; row < 16; ++row) {
      int b0 = lds_best[row * 3 + 0];
      int b1 = lds_best[row * 3 + 1];
      int b2 = lds_best[row * 3 + 2];
      const f32x4* m0 = (const f32x4*)(mem + ((size_t)(b0 * 3 + 0) << 10));
      const f32x4* m1 = (const f32x4*)(mem + ((size_t)(b1 * 3 + 1) << 10));
      const f32x4* m2 = (const f32x4*)(mem + ((size_t)(b2 * 3 + 2) << 10));
      f32x4 r = m0[g] + m1[g] + m2[g];
      *(f32x4*)(out + ((size_t)(rbase + row) << 10) + (g << 2)) = r;
    }
  }
}

extern "C" void kernel_launch(void* const* d_in, const int* in_sizes, int n_in,
                              void* d_out, int out_size, void* d_ws, size_t ws_size,
                              hipStream_t stream) {
  const float* x   = (const float*)d_in[0];   // (8, 4096, 1024) f32
  const float* mem = (const float*)d_in[1];   // (256, 3, 1024) f32
  const float* pb  = (const float*)d_in[2];   // (256, 3) f32
  float* out = (float*)d_out;                 // (8, 4096, 1024) f32
  unsigned short* Bmat = (unsigned short*)d_ws;  // 768*1024 bf16 = 1.5 MB scratch

  pnm_convert<<<3072, 256, 0, stream>>>(mem, Bmat);
  pnm_main<<<1024, 256, 0, stream>>>(x, mem, pb, Bmat, out);
}
